// Block_44238163149168
// MI455X (gfx1250) — matrix operand order repaired and checked by reading, not compiled, not run on hardware
//
#include <hip/hip_runtime.h>

// ---------------------------------------------------------------------------
// Types for CDNA5 WMMA (wave32): D(16x16 f32) = A(16x32 bf16) x B(32x16 bf16) + C
// ---------------------------------------------------------------------------
typedef __bf16 bf16;
typedef __attribute__((ext_vector_type(16))) __bf16 v16bf;
typedef __attribute__((ext_vector_type(8)))  __bf16 v8bf;
typedef __attribute__((ext_vector_type(8)))  float  v8f;

constexpr int BB  = 2;
constexpr int TT  = 2048;
constexpr int DD  = 1024;
constexpr int HH  = 16;
constexpr int HS  = 64;
constexpr int MM  = BB * TT;   // 4096 token rows
constexpr int FF  = 4 * DD;    // 4096 FFN hidden
constexpr int NQKV = 3 * HH * HS; // 3072 fused QKV output columns

__device__ __forceinline__ v8f vzero8() {
  v8f z;
#pragma unroll
  for (int i = 0; i < 8; ++i) z[i] = 0.0f;
  return z;
}

// Load a 16-element bf16 fragment as two contiguous 16B (8-elem) runs.
__device__ __forceinline__ v16bf ldfrag(const bf16* p0, const bf16* p1) {
  v8bf lo = *(const v8bf*)p0;
  v8bf hi = *(const v8bf*)p1;
  v16bf r;
#pragma unroll
  for (int i = 0; i < 8; ++i) { r[i] = lo[i]; r[i + 8] = hi[i]; }
  return r;
}

__device__ __forceinline__ v8f wmma_bf16(v16bf a, v16bf b, v8f c) {
  return __builtin_amdgcn_wmma_f32_16x16x32_bf16(false, a, false, b, (short)0, c,
                                                 false, false);
}

// Async copy of one 16B chunk global -> LDS (gfx1250 ASYNCcnt path).
__device__ __forceinline__ void async_cp16(const bf16* gptr, bf16* lptr) {
  unsigned lds_off = (unsigned)(size_t)(const void*)lptr; // low 32b = LDS offset
  asm volatile("global_load_async_to_lds_b128 %0, %1, off"
               :: "v"(lds_off), "v"(gptr) : "memory");
}

// ---------------------------------------------------------------------------
// Generic 128x128-tile GEMM: C = A[M,K] * Bt[N,K]^T, double-buffered async LDS.
// Block = 256 threads (8 waves); wave w computes rows [w*16, w*16+16) x 128 cols.
// mode 0: fused QKV scatter (q,k bf16 [B,H,T,HS]; v transposed [B,H,HS,T])
// mode 1: bias + relu -> bf16  (FFN1)
// mode 2: bias + residual -> f32 (FFN2)
// ---------------------------------------------------------------------------
__global__ __launch_bounds__(256) void k_gemm128(
    const bf16* __restrict__ A, const bf16* __restrict__ Bt, int K, int mode,
    const float* __restrict__ bias, const float* __restrict__ res,
    float* __restrict__ outF, bf16* __restrict__ outH,
    bf16* __restrict__ outQ, bf16* __restrict__ outK, bf16* __restrict__ outV) {
  __shared__ __align__(16) bf16 sA[2][128 * 32];  // 8 KB per buffer
  __shared__ __align__(16) bf16 sB[2][128 * 32];

  const int tid  = threadIdx.x;
  const int wid  = tid >> 5;
  const int lane = tid & 31;
  const int m    = lane & 15;
  const int half = lane >> 4;
  const int mrow0 = blockIdx.x * 128;
  const int nrow0 = blockIdx.y * 128;

  v8f acc[8];
#pragma unroll
  for (int nt = 0; nt < 8; ++nt) acc[nt] = vzero8();

  // Cooperative staging: 512 16B chunks per 128x32 panel; 2 A + 2 B per thread.
  auto stage = [&](int kb, int buf) {
#pragma unroll
    for (int j = 0; j < 2; ++j) {
      int c   = tid + 256 * j;          // 0..511
      int row = c >> 2;                 // 0..127
      int kc  = (c & 3) * 8;            // bf16 offset of 16B chunk within 32-K
      async_cp16(A  + (size_t)(mrow0 + row) * K + kb + kc, &sA[buf][c * 8]);
      async_cp16(Bt + (size_t)(nrow0 + row) * K + kb + kc, &sB[buf][c * 8]);
    }
  };

  const int K32 = K >> 5;
  stage(0, 0);
  for (int it = 0; it < K32; ++it) {
    const int cur = it & 1;
    if (it + 1 < K32) {
      stage((it + 1) * 32, cur ^ 1);                    // overlap next panel DMA
      asm volatile("s_wait_asynccnt 4" ::: "memory");   // current panel landed
    } else {
      asm volatile("s_wait_asynccnt 0" ::: "memory");
    }
    __syncthreads();

    // Register-stage ALL fragments first (one ds clause, one dscnt wait) ...
    const bf16* arow = &sA[cur][(wid * 16 + m) * 32 + half * 8];
    v16bf a = ldfrag(arow, arow + 16);
    v16bf bfr[8];
#pragma unroll
    for (int nt = 0; nt < 8; ++nt) {
      const bf16* brow = &sB[cur][(nt * 16 + m) * 32 + half * 8];
      bfr[nt] = ldfrag(brow, brow + 16);
    }
    // Hard scheduling fence: keep all ds_loads above, all WMMAs below, so the
    // 8 matrix ops issue back-to-back behind a single s_wait_dscnt.
    __builtin_amdgcn_sched_barrier(0);
#pragma unroll
    for (int nt = 0; nt < 8; ++nt) acc[nt] = wmma_bf16(a, bfr[nt], acc[nt]);

    __syncthreads();  // all reads done before this buffer is overwritten
  }

  // Epilogue. C layout: VGPR r -> row half*8+r, column = lane&15.
#pragma unroll
  for (int nt = 0; nt < 8; ++nt) {
#pragma unroll
    for (int r = 0; r < 8; ++r) {
      const int row = mrow0 + wid * 16 + half * 8 + r;
      const int col = nrow0 + nt * 16 + m;
      const float v = acc[nt][r];
      if (mode == 1) {
        outH[(size_t)row * FF + col] = (bf16)fmaxf(v + bias[col], 0.0f);
      } else if (mode == 2) {
        const size_t idx = (size_t)row * DD + col;
        outF[idx] = v + bias[col] + res[idx];
      } else {
        const int z = col >> 10, h = (col >> 6) & 15, hs = col & 63;
        const int b = row >> 11, t = row & (TT - 1);
        const bf16 val = (bf16)v;
        if (z == 0)      outQ[((size_t)(b * HH + h) * TT + t) * HS + hs] = val;
        else if (z == 1) outK[((size_t)(b * HH + h) * TT + t) * HS + hs] = val;
        else             outV[((size_t)(b * HH + h) * HS + hs) * TT + t] = val;
      }
    }
  }
}

// ---------------------------------------------------------------------------
// Weight conversion kernels (fp32 -> bf16, pre-transposed to [N,K])
// ---------------------------------------------------------------------------
// W [H, D, HS] -> Wt [H, HS, D]
__global__ void k_conv_qkvw(const float* __restrict__ w, bf16* __restrict__ wt) {
  size_t i = (size_t)blockIdx.x * 256 + threadIdx.x;
  if (i >= (size_t)HH * DD * HS) return;
  int h   = (int)(i / (DD * HS));
  int rem = (int)(i % (DD * HS));
  int d   = rem / HS;
  int e   = rem % HS;
  wt[((size_t)h * HS + e) * DD + d] = (bf16)w[i];
}

// in [R, C] f32 -> out [C, R] bf16
__global__ void k_transpose(const float* __restrict__ in, bf16* __restrict__ out,
                            int R, int C) {
  size_t i = (size_t)blockIdx.x * 256 + threadIdx.x;
  if (i >= (size_t)R * C) return;
  int r = (int)(i / C);
  int c = (int)(i % C);
  out[(size_t)c * R + r] = (bf16)in[i];
}

// ---------------------------------------------------------------------------
// LayerNorm kernels (one 256-thread block per row; D = 1024 -> 4 elems/thread)
// ---------------------------------------------------------------------------
__device__ __forceinline__ float block_sum(float v, float* sbuf) {
#pragma unroll
  for (int msk = 16; msk; msk >>= 1) v += __shfl_xor(v, msk);
  const int wid  = threadIdx.x >> 5;
  const int lane = threadIdx.x & 31;
  __syncthreads();
  if (lane == 0) sbuf[wid] = v;
  __syncthreads();
  float r = 0.0f;
#pragma unroll
  for (int i = 0; i < 8; ++i) r += sbuf[i];
  return r;
}

__global__ __launch_bounds__(256) void k_ln1(const float* __restrict__ e,
                                             const float* __restrict__ g,
                                             const float* __restrict__ bta,
                                             bf16* __restrict__ xln) {
  __shared__ float sbuf[8];
  const int row = blockIdx.x;
  const float* xr = e + (size_t)row * DD;
  float lv[4];
  float s = 0.0f, ss = 0.0f;
#pragma unroll
  for (int j = 0; j < 4; ++j) {
    int i = threadIdx.x + j * 256;
    float v = xr[i];
    lv[j] = v; s += v; ss += v * v;
  }
  s  = block_sum(s, sbuf);
  ss = block_sum(ss, sbuf);
  const float mean = s * (1.0f / DD);
  const float var  = ss * (1.0f / DD) - mean * mean;
  const float rstd = rsqrtf(var + 1e-5f);
#pragma unroll
  for (int j = 0; j < 4; ++j) {
    int i = threadIdx.x + j * 256;
    xln[(size_t)row * DD + i] = (bf16)((lv[j] - mean) * rstd * g[i] + bta[i]);
  }
}

// residual add (embds + attn) -> xres (f32) and LN2 -> y (bf16)
__global__ __launch_bounds__(256) void k_ln2(const float* __restrict__ e,
                                             const float* __restrict__ attn,
                                             const float* __restrict__ g,
                                             const float* __restrict__ bta,
                                             float* __restrict__ xres,
                                             bf16* __restrict__ y) {
  __shared__ float sbuf[8];
  const int row = blockIdx.x;
  const float* er = e    + (size_t)row * DD;
  const float* ar = attn + (size_t)row * DD;
  float lv[4];
  float s = 0.0f, ss = 0.0f;
#pragma unroll
  for (int j = 0; j < 4; ++j) {
    int i = threadIdx.x + j * 256;
    float v = er[i] + ar[i];
    xres[(size_t)row * DD + i] = v;
    lv[j] = v; s += v; ss += v * v;
  }
  s  = block_sum(s, sbuf);
  ss = block_sum(ss, sbuf);
  const float mean = s * (1.0f / DD);
  const float var  = ss * (1.0f / DD) - mean * mean;
  const float rstd = rsqrtf(var + 1e-5f);
#pragma unroll
  for (int j = 0; j < 4; ++j) {
    int i = threadIdx.x + j * 256;
    y[(size_t)row * DD + i] = (bf16)((lv[j] - mean) * rstd * g[i] + bta[i]);
  }
}

// ---------------------------------------------------------------------------
// Flash-style causal attention. One wave per 16-row query tile per (b,h).
// grid = (T/16/4, B*H), block = 128 (4 waves). Online softmax, f32 state.
// score scale = D^-0.5 = 1/32 (matches reference).
// ---------------------------------------------------------------------------
__global__ __launch_bounds__(128) void k_attn(const bf16* __restrict__ q,
                                              const bf16* __restrict__ k,
                                              const bf16* __restrict__ vT,
                                              float* __restrict__ attnout) {
  __shared__ __align__(16) bf16 psh[4][16 * 32];  // P tile per wave (1 KB)
  const int wid  = threadIdx.x >> 5;
  const int lane = threadIdx.x & 31;
  const int m    = lane & 15;
  const int half = lane >> 4;
  const int qt   = blockIdx.x * 4 + wid;          // 0..127
  const int bh   = blockIdx.y;                    // 0..31
  const bf16* Q  = q  + (size_t)bh * TT * HS;
  const bf16* Km = k  + (size_t)bh * TT * HS;
  const bf16* Vt = vT + (size_t)bh * HS * TT;

  const bf16* qrow = Q + (size_t)(qt * 16 + m) * HS + half * 8;
  const v16bf aq0 = ldfrag(qrow,      qrow + 16);   // K = 0..31
  const v16bf aq1 = ldfrag(qrow + 32, qrow + 48);   // K = 32..63

  const float NEGINF = -__builtin_inff();
  float mrow[8], lrow[8];
  v8f o[4];
#pragma unroll
  for (int r = 0; r < 8; ++r) { mrow[r] = NEGINF; lrow[r] = 0.0f; }
#pragma unroll
  for (int nt = 0; nt < 4; ++nt) o[nt] = vzero8();

  const float scale = 0.03125f;                   // 1/sqrt(1024)
  const int ktmax = (qt * 16 + 15) >> 5;          // inclusive, 32-key tiles

  for (int kt = 0; kt <= ktmax; ++kt) {
    const int kb = kt * 32;

    // Load all K fragments first, then 4 back-to-back score WMMAs.
    const bf16* kr0 = Km + (size_t)(kb + m) * HS + half * 8;
    const bf16* kr1 = Km + (size_t)(kb + 16 + m) * HS + half * 8;
    v16bf b0a = ldfrag(kr0,      kr0 + 16);
    v16bf b0b = ldfrag(kr0 + 32, kr0 + 48);
    v16bf b1a = ldfrag(kr1,      kr1 + 16);
    v16bf b1b = ldfrag(kr1 + 32, kr1 + 48);
    __builtin_amdgcn_sched_barrier(0);
    v8f s0 = vzero8(), s1 = vzero8();
    s0 = wmma_bf16(aq0, b0a, s0);
    s0 = wmma_bf16(aq1, b0b, s0);
    s1 = wmma_bf16(aq0, b1a, s1);
    s1 = wmma_bf16(aq1, b1b, s1);

    // Prefetch V fragments (independent of the softmax / P roundtrip).
    v16bf bv[4];
#pragma unroll
    for (int nt = 0; nt < 4; ++nt) {
      const bf16* vrow = Vt + (size_t)(nt * 16 + m) * TT + kb + half * 8;
      bv[nt] = ldfrag(vrow, vrow + 16);
    }

    // Causal mask + online softmax (row r lives in lanes of this half).
#pragma unroll
    for (int r = 0; r < 8; ++r) {
      const int row = qt * 16 + half * 8 + r;
      const int c0  = kb + m;
      const int c1  = kb + 16 + m;
      float x0 = (c0 <= row) ? s0[r] * scale : NEGINF;
      float x1 = (c1 <= row) ? s1[r] * scale : NEGINF;
      float mx = fmaxf(x0, x1);
#pragma unroll
      for (int msk = 8; msk; msk >>= 1) mx = fmaxf(mx, __shfl_xor(mx, msk));
      const float mnew  = fmaxf(mrow[r], mx);
      const float alpha = __expf(mrow[r] - mnew);
      const float p0 = __expf(x0 - mnew);
      const float p1 = __expf(x1 - mnew);
      float rs = p0 + p1;
#pragma unroll
      for (int msk = 8; msk; msk >>= 1) rs += __shfl_xor(rs, msk);
      lrow[r] = lrow[r] * alpha + rs;
      mrow[r] = mnew;
#pragma unroll
      for (int nt = 0; nt < 4; ++nt) o[nt][r] *= alpha;
      psh[wid][(half * 8 + r) * 32 + m]      = (bf16)p0;
      psh[wid][(half * 8 + r) * 32 + 16 + m] = (bf16)p1;
    }

    // P (C-layout) -> LDS -> A-layout, then 4 back-to-back PV WMMAs.
    const bf16* prow = &psh[wid][m * 32 + half * 8];
    v16bf ap = ldfrag(prow, prow + 16);
    __builtin_amdgcn_sched_barrier(0);
#pragma unroll
    for (int nt = 0; nt < 4; ++nt) o[nt] = wmma_bf16(ap, bv[nt], o[nt]);
  }

  const int b = bh >> 4, h = bh & 15;
#pragma unroll
  for (int r = 0; r < 8; ++r) {
    const int t = qt * 16 + half * 8 + r;
    const float inv = 1.0f / lrow[r];
    float* orow = attnout + ((size_t)b * TT + t) * DD + h * HS;
#pragma unroll
    for (int nt = 0; nt < 4; ++nt) orow[nt * 16 + m] = o[nt][r] * inv;
  }
}

// ---------------------------------------------------------------------------
// Host-side orchestration
// ---------------------------------------------------------------------------
extern "C" void kernel_launch(void* const* d_in, const int* in_sizes, int n_in,
                              void* d_out, int out_size, void* d_ws, size_t ws_size,
                              hipStream_t stream) {
  const float* embds = (const float*)d_in[0];
  const float* Wq    = (const float*)d_in[1];
  const float* Wk    = (const float*)d_in[2];
  const float* Wv    = (const float*)d_in[3];
  const float* ln1_g = (const float*)d_in[4];
  const float* ln1_b = (const float*)d_in[5];
  const float* ln2_g = (const float*)d_in[6];
  const float* ln2_b = (const float*)d_in[7];
  const float* W1    = (const float*)d_in[8];
  const float* b1    = (const float*)d_in[9];
  const float* W2    = (const float*)d_in[10];
  const float* b2    = (const float*)d_in[11];
  float* out = (float*)d_out;

  char* ws = (char*)d_ws;
  size_t off = 0;
  auto alloc = [&](size_t bytes) -> char* {
    char* p = ws + off;
    off += (bytes + 255) & ~(size_t)255;
    return p;
  };

  bf16* xln   = (bf16*)alloc((size_t)MM * DD * 2);            // 8 MB
  bf16* wAllT = (bf16*)alloc((size_t)NQKV * DD * 2);          // 6 MB (Wq|Wk|Wv)^T
  bf16* qb    = (bf16*)alloc((size_t)BB * HH * TT * HS * 2);  // 8 MB
  bf16* kb    = (bf16*)alloc((size_t)BB * HH * TT * HS * 2);
  bf16* vTb   = (bf16*)alloc((size_t)BB * HH * HS * TT * 2);
  float* attn = (float*)alloc((size_t)MM * DD * 4);           // 16 MB
  float* xres = (float*)alloc((size_t)MM * DD * 4);           // 16 MB
  bf16* yb    = (bf16*)alloc((size_t)MM * DD * 2);            // 8 MB
  bf16* w1T   = (bf16*)alloc((size_t)FF * DD * 2);            // 8 MB
  bf16* w2T   = (bf16*)alloc((size_t)DD * FF * 2);            // 8 MB
  bf16* hb    = (bf16*)alloc((size_t)MM * FF * 2);            // 32 MB
  (void)ws_size; (void)n_in; (void)in_sizes; (void)out_size;

  // 1) weight conversions (bf16, pre-transposed to [N,K]); QKV concatenated
  {
    int blocks = (HH * DD * HS + 255) / 256;   // 4096
    k_conv_qkvw<<<blocks, 256, 0, stream>>>(Wq, wAllT + 0 * (size_t)HH * HS * DD);
    k_conv_qkvw<<<blocks, 256, 0, stream>>>(Wk, wAllT + 1 * (size_t)HH * HS * DD);
    k_conv_qkvw<<<blocks, 256, 0, stream>>>(Wv, wAllT + 2 * (size_t)HH * HS * DD);
    int tblocks = (DD * FF + 255) / 256;       // 16384
    k_transpose<<<tblocks, 256, 0, stream>>>(W1, w1T, DD, FF);  // [D,4D]->[4D,D]
    k_transpose<<<tblocks, 256, 0, stream>>>(W2, w2T, FF, DD);  // [4D,D]->[D,4D]
  }

  // 2) LN1
  k_ln1<<<MM, 256, 0, stream>>>(embds, ln1_g, ln1_b, xln);

  // 3) fused QKV projection: [4096,1024] x [1024,3072] (mode 0)
  k_gemm128<<<dim3(MM / 128, NQKV / 128), 256, 0, stream>>>(
      xln, wAllT, DD, 0, nullptr, nullptr, nullptr, nullptr, qb, kb, vTb);

  // 4) causal flash attention (WMMA)
  k_attn<<<dim3(TT / 16 / 4, BB * HH), 128, 0, stream>>>(qb, kb, vTb, attn);

  // 5) residual + LN2
  k_ln2<<<MM, 256, 0, stream>>>(embds, attn, ln2_g, ln2_b, xres, yb);

  // 6) FFN1: relu(y @ W1 + b1) (mode 1)
  k_gemm128<<<dim3(MM / 128, FF / 128), 256, 0, stream>>>(
      yb, w1T, DD, 1, b1, nullptr, nullptr, hb, nullptr, nullptr, nullptr);

  // 7) FFN2: hb @ W2 + b2 + xres (mode 2)
  k_gemm128<<<dim3(MM / 128, DD / 128), 256, 0, stream>>>(
      hb, w2T, FF, 2, b2, xres, out, nullptr, nullptr, nullptr, nullptr);
}
